// DIMKT_25254407701101
// MI455X (gfx1250) — compile-verified, matrix operand order checked by reading, not run on hardware
//
#include <hip/hip_runtime.h>
#include <hip/hip_bf16.h>

// ---------------------------------------------------------------------------
// DIMKT on gfx1250 (MI455X).
//   B=128, L=512, E=128.  All GEMMs via v_wmma_f32_16x16x32_f16 (wave32).
//
//   Stage 0: gather embeddings -> f16 concat buffers.
//   Stage 1: big parallel WMMA GEMMs for every state-independent term:
//            X  = cat(q,c,sd,qd)@W1+b1            (f16)
//            U2 = X@W2+b2, U3 = X@W3+b3           (f32)
//            A4 = a@W4[E:2E]+b4, A5 = a@W5[E:2E]+b5
//            C6 = a@W6[E:2E]+sd@W6[2E:3E]+qd@W6[3E:4E]+b6
//   Stage 2: sequential scan, 8 WGs x 16 batch rows, weights in LDS.
//            Recurrent terms are WMMAs whose C accumulator is initialized
//            from the precomputed tensors (W2/W3 stored negated so
//            U2 - k@W2 is a single accumulating WMMA).  The per-step
//            precomputed stripes are streamed into LDS one step ahead by
//            the Tensor Data Mover (TENSOR_LOAD_TO_LDS + s_wait_tensorcnt),
//            double-buffered, so the serial chain never waits on HBM.
// ---------------------------------------------------------------------------

typedef __attribute__((ext_vector_type(16))) _Float16 v16h;
typedef __attribute__((ext_vector_type(8)))  float    v8f;
typedef __attribute__((ext_vector_type(4)))  unsigned int v4u;
typedef __attribute__((ext_vector_type(4)))  unsigned int u32x4;
typedef __attribute__((ext_vector_type(8)))  int i32x8;
typedef __attribute__((ext_vector_type(4)))  int i32x4;

#define B_    128
#define L_    512
#define E_    128
#define M_TOT (B_ * L_)   // 65536

// LDS map for the scan kernel (bytes)
#define LDS_KF16   0        // 16x128 f16   (4 KB)
#define LDS_KF32   4096     // 16x128 f32   (8 KB)
#define LDS_SDF    12288    // 16x128 f16   (4 KB)
#define LDS_W      16384    // 167936 B of f16 weights
#define LDS_STAGE  184320   // 2 x 5 x 8192 B TDM staging (80 KB)
#define LDS_TOTAL  266240

union HFrag { v16h h; v4u u[2]; };

// 16-bit A/B fragment loader (16x32 tile), row-major [rows][ld] f16 source.
// Lane layout per CDNA5 ISA: lanes 0-15 hold K={kb*32+0..7, +16..23},
// lanes 16-31 hold K={kb*32+8..15, +24..31}; row = lane&15 (M for A, N for B^T).
__device__ __forceinline__ v16h load_frag16(const _Float16* base, int row,
                                            int ld, int kb, int hi) {
  const _Float16* p = base + (size_t)row * ld + kb * 32 + hi * 8;
  HFrag f;
  f.u[0] = *(const v4u*)p;
  f.u[1] = *(const v4u*)(p + 16);
  return f.h;
}

__device__ __forceinline__ v8f wmma_f16(v16h a, v16h b, v8f c) {
  return __builtin_amdgcn_wmma_f32_16x16x32_f16(false, a, false, b,
                                                (short)0, c, false, false);
}

__device__ __forceinline__ float sigf(float x) {
  return 1.0f / (1.0f + __expf(-x));
}
__device__ __forceinline__ float tanhf_(float x) {
  float ax = fabsf(x);
  float e  = __expf(-2.0f * ax);
  float t  = (1.0f - e) / (1.0f + e);
  return copysignf(t, x);
}

// ---------------------------------------------------------------------------
// Tensor Data Mover: 1-D contiguous global->LDS DMA of nelem f32 elements.
// D# packing per CDNA5 ISA 8.3/8.4:
//   group0: [1:0]=count=1 | lds_addr | global_addr[56:0] | type=2
//   group1: data_size=2 (4B), tensor_dim0=tile_dim0=nelem, tensor_dim1=1
//   groups 2/3 (+extra group word block): zero (tensor <= 2D)
// Toolchain uses the 6-arg builtin form (g0, g1, g2, g3, g4, cpol).
// ---------------------------------------------------------------------------
__device__ __forceinline__ void tdm_load_1d(const void* gsrc, unsigned lds_off,
                                            unsigned nelem) {
  unsigned long long ga = (unsigned long long)(uintptr_t)gsrc;
  u32x4 g0;
  g0[0] = 1u;                                               // count=1
  g0[1] = lds_off;                                          // LDS byte addr
  g0[2] = (unsigned)ga;                                     // global addr lo
  g0[3] = (unsigned)((ga >> 32) & 0x1FFFFFFu) | 0x80000000u; // addr hi | type=2
  i32x8 g1;
  g1[0] = 0x00020000;                              // data_size = 4 bytes
  g1[1] = (int)((nelem & 0xFFFFu) << 16);          // tensor_dim0[15:0]
  g1[2] = (int)((nelem >> 16) | (1u << 16));       // tensor_dim0[31:16] | tdim1=1
  g1[3] = (int)((nelem & 0xFFFFu) << 16);          // tile_dim0
  g1[4] = 0;                                       // tile_dim1/2 unused
  g1[5] = (int)nelem;                              // tensor_dim0_stride lo
  g1[6] = 0;
  g1[7] = 0;
  i32x4 z4 = {0, 0, 0, 0};
  i32x8 z8 = {0, 0, 0, 0, 0, 0, 0, 0};
  __builtin_amdgcn_tensor_load_to_lds(g0, g1, z4, z4, z8, 0);
}

// ---------------------------------------------------------------------------
// Stage 0: embedding gather -> f16.  One thread per (m, e).
// m = l*B + b (step-major rows so per-step slices are contiguous later).
// ---------------------------------------------------------------------------
__global__ __launch_bounds__(256) void gather_k(
    const int* __restrict__ q, const int* __restrict__ c,
    const int* __restrict__ sd, const int* __restrict__ qd,
    const int* __restrict__ a,
    const float* __restrict__ q_tab, const float* __restrict__ c_tab,
    const float* __restrict__ sd_tab, const float* __restrict__ qd_tab,
    const float* __restrict__ a_tab,
    _Float16* __restrict__ Ecat,   // [M_TOT][4E] = [q|c|sd|qd]
    _Float16* __restrict__ Af16)   // [M_TOT][E]
{
  int t = blockIdx.x * 256 + threadIdx.x;   // < M_TOT * 128
  int m = t >> 7, e = t & 127;
  int b = m & (B_ - 1), l = m >> 7;
  int base = b * L_ + l;
  int qi = q[base], ci = c[base], sdi = sd[base], qdi = qd[base], ai = a[base];
  size_t mo = (size_t)m * (4 * E_);
  Ecat[mo + e]           = (_Float16)q_tab[(size_t)qi * E_ + e];
  Ecat[mo + E_ + e]      = (_Float16)c_tab[(size_t)ci * E_ + e];
  Ecat[mo + 2 * E_ + e]  = (_Float16)sd_tab[(size_t)sdi * E_ + e];
  Ecat[mo + 3 * E_ + e]  = (_Float16)qd_tab[(size_t)qdi * E_ + e];
  Af16[(size_t)m * E_ + e] = (_Float16)a_tab[(size_t)ai * E_ + e];
}

// ---------------------------------------------------------------------------
// Weight prep: f16 transposed weights.  seqw = [W2tn|W3tn|W4at|W5at|W6at|Wft]
// (contiguous so the scan kernel does a single LDS copy).
// ---------------------------------------------------------------------------
__global__ __launch_bounds__(256) void wprep_k(
    const float* __restrict__ W1, const float* __restrict__ W2,
    const float* __restrict__ W3, const float* __restrict__ W4,
    const float* __restrict__ W5, const float* __restrict__ W6,
    const float* __restrict__ Wf,
    _Float16* __restrict__ W1t,
    _Float16* __restrict__ W2t,  _Float16* __restrict__ W3t,
    _Float16* __restrict__ W4bt, _Float16* __restrict__ W5bt,
    _Float16* __restrict__ W6b1t, _Float16* __restrict__ W6b2t,
    _Float16* __restrict__ W6b3t,
    _Float16* __restrict__ seqw)
{
  _Float16* W2tn = seqw;
  _Float16* W3tn = seqw + 16384;
  _Float16* W4at = seqw + 32768;
  _Float16* W5at = seqw + 49152;
  _Float16* W6at = seqw + 65536;
  _Float16* Wft  = seqw + 81920;

  int t = blockIdx.x * 256 + threadIdx.x;
  if (t < 65536) {                       // W1t [128][512]
    int n = t >> 9, k = t & 511;
    W1t[t] = (_Float16)W1[k * E_ + n];
    return;
  }
  t -= 65536;
  if (t < 16384 * 12) {
    int which = t >> 14;
    int idx = t & 16383;
    int n = idx >> 7, k = idx & 127;
    switch (which) {
      case 0:  W2t[idx]   = (_Float16)W2[k * E_ + n]; break;
      case 1:  W3t[idx]   = (_Float16)W3[k * E_ + n]; break;
      case 2:  W2tn[idx]  = (_Float16)(-W2[k * E_ + n]); break;
      case 3:  W3tn[idx]  = (_Float16)(-W3[k * E_ + n]); break;
      case 4:  W4at[idx]  = (_Float16)W4[k * E_ + n]; break;
      case 5:  W4bt[idx]  = (_Float16)W4[(128 + k) * E_ + n]; break;
      case 6:  W5at[idx]  = (_Float16)W5[k * E_ + n]; break;
      case 7:  W5bt[idx]  = (_Float16)W5[(128 + k) * E_ + n]; break;
      case 8:  W6at[idx]  = (_Float16)W6[k * E_ + n]; break;
      case 9:  W6b1t[idx] = (_Float16)W6[(128 + k) * E_ + n]; break;
      case 10: W6b2t[idx] = (_Float16)W6[(256 + k) * E_ + n]; break;
      case 11: W6b3t[idx] = (_Float16)W6[(384 + k) * E_ + n]; break;
    }
    return;
  }
  t -= 16384 * 12;
  if (t < 2048) {                        // Wft [16][128], rows 10..15 zero
    int n = t >> 7, k = t & 127;
    Wft[t] = (n < 10) ? (_Float16)Wf[k * 10 + n] : (_Float16)0.0f;
  }
}

// ---------------------------------------------------------------------------
// Stage 1: generic WMMA GEMM.  C[M x 128] = A[M x K](f16) * Wt[128 x K]^T.
// Block = 256 threads = 8 waves; block does 128 rows, wave w rows 16w..16w+15,
// each wave accumulates all 8 column tiles so A fragments are loaded once/kb.
// accumulate=0: acc = bias[col];  accumulate=1: acc = Cf32 (read-mod-write).
// Cf16 != null -> store f16, else store f32.
// ---------------------------------------------------------------------------
__global__ __launch_bounds__(256) void gemm_k(
    const _Float16* __restrict__ A, int lda,
    const _Float16* __restrict__ Wt, int K,
    const float* __restrict__ bias, int accumulate,
    float* __restrict__ Cf32, _Float16* __restrict__ Cf16)
{
  const int wave  = threadIdx.x >> 5;
  const int lane  = threadIdx.x & 31;
  const int laneM = lane & 15;
  const int hi    = lane >> 4;
  const long rowBase = (long)blockIdx.x * 128 + wave * 16;

  v8f acc[8];
#pragma unroll
  for (int tn = 0; tn < 8; ++tn) {
    int col = tn * 16 + laneM;
    if (accumulate) {
#pragma unroll
      for (int r = 0; r < 8; ++r)
        acc[tn][r] = Cf32[(size_t)(rowBase + r + hi * 8) * E_ + col];
    } else {
      float bv = bias[col];
#pragma unroll
      for (int r = 0; r < 8; ++r) acc[tn][r] = bv;
    }
  }

  const int nkb = K >> 5;
  for (int kb = 0; kb < nkb; ++kb) {
    v16h aF = load_frag16(A, (int)rowBase + laneM, lda, kb, hi);
#pragma unroll
    for (int tn = 0; tn < 8; ++tn) {
      v16h bF = load_frag16(Wt, tn * 16 + laneM, K, kb, hi);
      acc[tn] = wmma_f16(aF, bF, acc[tn]);
    }
  }

#pragma unroll
  for (int tn = 0; tn < 8; ++tn) {
    int col = tn * 16 + laneM;
    if (Cf16) {
#pragma unroll
      for (int r = 0; r < 8; ++r)
        Cf16[(size_t)(rowBase + r + hi * 8) * E_ + col] = (_Float16)acc[tn][r];
    } else {
#pragma unroll
      for (int r = 0; r < 8; ++r)
        Cf32[(size_t)(rowBase + r + hi * 8) * E_ + col] = acc[tn][r];
    }
  }
}

// ---------------------------------------------------------------------------
// Stage 2: sequential scan.  8 blocks x 256 threads (8 waves).
// Block bg owns batch rows [bg*16, bg*16+16).  Wave w owns E-column tile w.
// Per-step precomputed stripes are TDM-prefetched into double-buffered LDS.
// ---------------------------------------------------------------------------
__global__ __launch_bounds__(256) void seq_k(
    const float* __restrict__ knowledge,
    const _Float16* __restrict__ seqw,    // [W2tn|W3tn|W4at|W5at|W6at|Wft]
    const float* __restrict__ U2, const float* __restrict__ U3,
    const float* __restrict__ A4, const float* __restrict__ A5,
    const float* __restrict__ C6,
    const float* __restrict__ bf,
    float* __restrict__ out)              // [B][L][10]
{
  extern __shared__ char smem[];
  _Float16* kf16 = (_Float16*)(smem + LDS_KF16);
  float*    kf32 = (float*)(smem + LDS_KF32);
  _Float16* sdf  = (_Float16*)(smem + LDS_SDF);
  _Float16* wS   = (_Float16*)(smem + LDS_W);
  _Float16* w2s = wS;                 // -W2^T  [128][128]
  _Float16* w3s = wS + 16384;         // -W3^T
  _Float16* w4s = wS + 32768;         //  W4a^T
  _Float16* w5s = wS + 49152;         //  W5a^T
  _Float16* w6s = wS + 65536;         //  W6a^T
  _Float16* wfs = wS + 81920;         //  Wf^T padded [16][128]

  const int tid = threadIdx.x;
  const int wave  = tid >> 5;
  const int lane  = tid & 31;
  const int laneM = lane & 15;
  const int hi    = lane >> 4;
  const int tn    = wave;                 // E-column tile
  const int col   = tn * 16 + laneM;
  const int bg    = blockIdx.x;           // batch tile
  const float bfv = (laneM < 10) ? bf[laneM] : 0.0f;

  // LDS byte address of the TDM staging area (low 32 bits of flat LDS addr).
  const unsigned stage_lds =
      (unsigned)(uintptr_t)(void*)(smem + LDS_STAGE);

  // cooperative weight copy: 167936 B = 10496 x 16 B
  {
    const v4u* src = (const v4u*)seqw;
    v4u* dst = (v4u*)wS;
    for (int i = tid; i < 10496; i += 256) dst[i] = src[i];
  }
  // k0 = broadcast(knowledge)
  for (int i = tid; i < 16 * E_; i += 256) {
    float v = knowledge[i & (E_ - 1)];
    kf32[i] = v;
    kf16[i] = (_Float16)v;
  }

  // TDM prologue: stream step 0's stripes into buffer 0.
  const size_t stripe0 = (size_t)bg * 16 * E_;   // per-step stripe offset
  if (wave == 0) {
    size_t go = stripe0;   // l = 0
    tdm_load_1d(U2 + go, stage_lds + 0 * 8192, 2048);
    tdm_load_1d(U3 + go, stage_lds + 1 * 8192, 2048);
    tdm_load_1d(A4 + go, stage_lds + 2 * 8192, 2048);
    tdm_load_1d(A5 + go, stage_lds + 3 * 8192, 2048);
    tdm_load_1d(C6 + go, stage_lds + 4 * 8192, 2048);
  }
  __syncthreads();

  for (int l = 0; l < L_; ++l) {
    // ---- TDM: issue step l+1 into the other buffer, wait for step l ----
    if (wave == 0) {
      if (l + 1 < L_) {
        size_t go = (size_t)(l + 1) * B_ * E_ + stripe0;
        unsigned sb = stage_lds + ((l + 1) & 1) * 40960;
        tdm_load_1d(U2 + go, sb + 0 * 8192, 2048);
        tdm_load_1d(U3 + go, sb + 1 * 8192, 2048);
        tdm_load_1d(A4 + go, sb + 2 * 8192, 2048);
        tdm_load_1d(A5 + go, sb + 3 * 8192, 2048);
        tdm_load_1d(C6 + go, sb + 4 * 8192, 2048);
        __builtin_amdgcn_s_wait_tensorcnt((short)5);  // step l resident
      } else {
        __builtin_amdgcn_s_wait_tensorcnt((short)0);
      }
    }
    __syncthreads();   // step-l staging visible to all waves

    const float* SU2 = (const float*)(smem + LDS_STAGE + (l & 1) * 40960);
    const float* SU3 = SU2 + 2048;
    const float* SA4 = SU2 + 4096;
    const float* SA5 = SU2 + 6144;
    const float* SC6 = SU2 + 8192;

    // A fragments of old k (local rows 0..15, K = 0..127)
    v16h aK[4];
#pragma unroll
    for (int kb = 0; kb < 4; ++kb) aK[kb] = load_frag16(kf16, laneM, E_, kb, hi);

    // ---- phase A: arg2 = U2 - k@W2, arg3 = U3 - k@W3, SDFt -> LDS ----
    v8f acc2, acc3;
#pragma unroll
    for (int r = 0; r < 8; ++r) {
      int li = (hi * 8 + r) * E_ + col;
      acc2[r] = SU2[li];
      acc3[r] = SU3[li];
    }
#pragma unroll
    for (int kb = 0; kb < 4; ++kb) {
      acc2 = wmma_f16(aK[kb], load_frag16(w2s, col, E_, kb, hi), acc2);
      acc3 = wmma_f16(aK[kb], load_frag16(w3s, col, E_, kb, hi), acc3);
    }
#pragma unroll
    for (int r = 0; r < 8; ++r) {
      float s = sigf(acc2[r]) * tanhf_(acc3[r]);
      sdf[(hi * 8 + r) * E_ + col] = (_Float16)s;
    }
    __syncthreads();   // SDFt visible to all waves; all aK loads complete

    // ---- phase B: PKAt, gate, state update ----
    v16h sK[4];
#pragma unroll
    for (int kb = 0; kb < 4; ++kb) sK[kb] = load_frag16(sdf, laneM, E_, kb, hi);

    v8f acc4, acc5, acc6;
#pragma unroll
    for (int r = 0; r < 8; ++r) {
      int li = (hi * 8 + r) * E_ + col;
      acc4[r] = SA4[li];
      acc5[r] = SA5[li];
      acc6[r] = SC6[li];
    }
#pragma unroll
    for (int kb = 0; kb < 4; ++kb) {
      acc4 = wmma_f16(sK[kb], load_frag16(w4s, col, E_, kb, hi), acc4);
      acc5 = wmma_f16(sK[kb], load_frag16(w5s, col, E_, kb, hi), acc5);
      acc6 = wmma_f16(aK[kb], load_frag16(w6s, col, E_, kb, hi), acc6);
    }
#pragma unroll
    for (int r = 0; r < 8; ++r) {
      float pk = sigf(acc4[r]) * tanhf_(acc5[r]);
      float g  = sigf(acc6[r]);
      int  li  = (hi * 8 + r) * E_ + col;
      float kn = g * kf32[li] + (1.0f - g) * pk;
      kf32[li] = kn;
      kf16[li] = (_Float16)kn;
    }
    __syncthreads();   // new k visible

    // ---- phase C: logits = k_new @ Wf + bf (wave 0 only; N padded to 16) ----
    if (wave == 0) {
      v16h nK[4];
#pragma unroll
      for (int kb = 0; kb < 4; ++kb)
        nK[kb] = load_frag16(kf16, laneM, E_, kb, hi);
      v8f accF;
#pragma unroll
      for (int r = 0; r < 8; ++r) accF[r] = bfv;
#pragma unroll
      for (int kb = 0; kb < 4; ++kb)
        accF = wmma_f16(nK[kb], load_frag16(wfs, laneM, E_, kb, hi), accF);
      if (laneM < 10) {
#pragma unroll
        for (int r = 0; r < 8; ++r) {
          int b = bg * 16 + hi * 8 + r;
          out[((size_t)b * L_ + l) * 10 + laneM] = sigf(accF[r]);
        }
      }
    }
  }
}

// ---------------------------------------------------------------------------
extern "C" void kernel_launch(void* const* d_in, const int* in_sizes, int n_in,
                              void* d_out, int out_size, void* d_ws,
                              size_t ws_size, hipStream_t stream)
{
  (void)in_sizes; (void)n_in; (void)out_size; (void)ws_size;
  const int* q  = (const int*)d_in[0];
  const int* c  = (const int*)d_in[1];
  const int* sd = (const int*)d_in[2];
  const int* qd = (const int*)d_in[3];
  const int* a  = (const int*)d_in[4];
  // d_in[5..8] = shifted variants: unused by the output.
  const float* knowledge = (const float*)d_in[9];
  const float* q_tab  = (const float*)d_in[10];
  const float* c_tab  = (const float*)d_in[11];
  const float* sd_tab = (const float*)d_in[12];
  const float* qd_tab = (const float*)d_in[13];
  const float* a_tab  = (const float*)d_in[14];
  const float* W1 = (const float*)d_in[15]; const float* b1 = (const float*)d_in[16];
  const float* W2 = (const float*)d_in[17]; const float* b2 = (const float*)d_in[18];
  const float* W3 = (const float*)d_in[19]; const float* b3 = (const float*)d_in[20];
  const float* W4 = (const float*)d_in[21]; const float* b4 = (const float*)d_in[22];
  const float* W5 = (const float*)d_in[23]; const float* b5 = (const float*)d_in[24];
  const float* W6 = (const float*)d_in[25]; const float* b6 = (const float*)d_in[26];
  const float* Wf = (const float*)d_in[27]; const float* bf = (const float*)d_in[28];
  float* out = (float*)d_out;

  char* ws = (char*)d_ws;
  size_t off = 0;
  auto take = [&](size_t bytes) -> char* {
    char* p = ws + off;
    off = (off + bytes + 255) & ~(size_t)255;
    return p;
  };

  _Float16* Ecat = (_Float16*)take((size_t)M_TOT * 4 * E_ * 2);  // 64 MB
  _Float16* Af16 = (_Float16*)take((size_t)M_TOT * E_ * 2);      // 16 MB
  _Float16* Xf16 = (_Float16*)take((size_t)M_TOT * E_ * 2);      // 16 MB
  float* U2 = (float*)take((size_t)M_TOT * E_ * 4);              // 32 MB each
  float* U3 = (float*)take((size_t)M_TOT * E_ * 4);
  float* A4 = (float*)take((size_t)M_TOT * E_ * 4);
  float* A5 = (float*)take((size_t)M_TOT * E_ * 4);
  float* C6 = (float*)take((size_t)M_TOT * E_ * 4);
  _Float16* W1t   = (_Float16*)take(512 * 128 * 2);
  _Float16* W2t   = (_Float16*)take(128 * 128 * 2);
  _Float16* W3t   = (_Float16*)take(128 * 128 * 2);
  _Float16* W4bt  = (_Float16*)take(128 * 128 * 2);
  _Float16* W5bt  = (_Float16*)take(128 * 128 * 2);
  _Float16* W6b1t = (_Float16*)take(128 * 128 * 2);
  _Float16* W6b2t = (_Float16*)take(128 * 128 * 2);
  _Float16* W6b3t = (_Float16*)take(128 * 128 * 2);
  _Float16* seqw  = (_Float16*)take(167936);   // [W2tn|W3tn|W4at|W5at|W6at|Wft]

  // Stage 0: gathers + weight prep
  gather_k<<<(M_TOT * 128) / 256, 256, 0, stream>>>(
      q, c, sd, qd, a, q_tab, c_tab, sd_tab, qd_tab, a_tab, Ecat, Af16);
  wprep_k<<<(65536 + 16384 * 12 + 2048) / 256, 256, 0, stream>>>(
      W1, W2, W3, W4, W5, W6, Wf,
      W1t, W2t, W3t, W4bt, W5bt, W6b1t, W6b2t, W6b3t, seqw);

  const int gblk = M_TOT / 128;   // 512 blocks
  // X = cat @ W1 + b1   (f16 out)
  gemm_k<<<gblk, 256, 0, stream>>>(Ecat, 512, W1t, 512, b1, 0, nullptr, Xf16);
  // U2 = X @ W2 + b2, U3 = X @ W3 + b3
  gemm_k<<<gblk, 256, 0, stream>>>(Xf16, 128, W2t, 128, b2, 0, U2, nullptr);
  gemm_k<<<gblk, 256, 0, stream>>>(Xf16, 128, W3t, 128, b3, 0, U3, nullptr);
  // A4 = a @ W4[E:2E] + b4, A5 = a @ W5[E:2E] + b5
  gemm_k<<<gblk, 256, 0, stream>>>(Af16, 128, W4bt, 128, b4, 0, A4, nullptr);
  gemm_k<<<gblk, 256, 0, stream>>>(Af16, 128, W5bt, 128, b5, 0, A5, nullptr);
  // C6 = a @ W6[E:2E] + b6, += sd @ W6[2E:3E], += qd @ W6[3E:4E]
  gemm_k<<<gblk, 256, 0, stream>>>(Af16, 128, W6b1t, 128, b6, 0, C6, nullptr);
  gemm_k<<<gblk, 256, 0, stream>>>(Ecat + 256, 512, W6b2t, 128, nullptr, 1, C6, nullptr);
  gemm_k<<<gblk, 256, 0, stream>>>(Ecat + 384, 512, W6b3t, 128, nullptr, 1, C6, nullptr);

  // Stage 2: sequential scan (8 batch tiles; double-buffered TDM staging)
  seq_k<<<8, 256, LDS_TOTAL, stream>>>(knowledge, seqw, U2, U3, A4, A5, C6,
                                       bf, out);
}
